// TobitKalmanLSTM_25031069401163
// MI455X (gfx1250) — matrix operand order repaired and checked by reading, not compiled, not run on hardware
//
#include <hip/hip_runtime.h>
#include <hip/hip_bf16.h>

// ---------------- problem dims ----------------
constexpr int T_STEPS = 512;
constexpr int OBSN    = 64;
constexpr int HIDN    = 1024;
constexpr int STAN    = 128;
constexpr int GATES   = 4 * HIDN;          // 4096
constexpr int KCAT    = STAN + HIDN;       // 1152
constexpr int GATE_CHUNKS = KCAT / 32;     // 36
constexpr int GATE_TILES  = GATES / 16;    // 256
constexpr int FC_CHUNKS   = HIDN / 32;     // 32
constexpr int FC_TILES    = STAN / 16;     // 8

constexpr int NB = 32;    // persistent blocks (must all be resident)
constexpr int NT = 512;   // threads per block (16 waves)
constexpr int NTHREADS = NB * NT;
constexpr int NWAVES   = NTHREADS / 32;

typedef __attribute__((ext_vector_type(16))) __bf16 v16bf;
typedef __attribute__((ext_vector_type(8)))  float  v8f;

// ---------------- workspace layout (bytes) ----------------
constexpr size_t GATE_PK_ELEMS = (size_t)GATE_TILES * GATE_CHUNKS * 512; // 4,718,592
constexpr size_t FC_PK_ELEMS   = (size_t)FC_TILES   * FC_CHUNKS   * 512; //   131,072

constexpr size_t O_PK_GM = 0;
constexpr size_t O_PK_GF = O_PK_GM + GATE_PK_ELEMS * 2;
constexpr size_t O_PK_GQ = O_PK_GF + GATE_PK_ELEMS * 2;
constexpr size_t O_PK_FM = O_PK_GQ + GATE_PK_ELEMS * 2;
constexpr size_t O_PK_FF = O_PK_FM + FC_PK_ELEMS * 2;
constexpr size_t O_PK_FQ = O_PK_FF + FC_PK_ELEMS * 2;
constexpr size_t O_F32   = O_PK_FQ + FC_PK_ELEMS * 2;   // multiple of 256

// float-indexed offsets inside the f32 region
constexpr size_t F_YMID = 0;
constexpr size_t F_HM   = F_YMID + 128;
constexpr size_t F_CM   = F_HM + 1024;
constexpr size_t F_HF   = F_CM + 1024;
constexpr size_t F_CF   = F_HF + 1024;
constexpr size_t F_HQ   = F_CF + 1024;
constexpr size_t F_CQ   = F_HQ + 1024;
constexpr size_t F_GM   = F_CQ + 1024;
constexpr size_t F_GF   = F_GM + 4096;
constexpr size_t F_GQ   = F_GF + 4096;
constexpr size_t F_FV   = F_GQ + 4096;
constexpr size_t F_QV   = F_FV + 128;
constexpr size_t F_P    = F_QV + 128;
constexpr size_t F_R1   = F_P + 16384;
constexpr size_t F_HP   = F_R1 + 8192;
constexpr size_t F_GJ   = F_HP + 8192;
constexpr size_t F_KB   = F_GJ + 12288;
constexpr size_t F_Z    = F_KB + 8192;
constexpr size_t F_PUN  = F_Z + 64;
constexpr size_t F_EY   = F_PUN + 64;
constexpr size_t F_WV   = F_EY + 64;
constexpr size_t F_END  = F_WV + 64;

constexpr size_t O_XHM  = O_F32 + ((F_END * 4 + 255) & ~size_t(255));
constexpr size_t O_XHF  = O_XHM + 2560;   // 1152 bf16 each, padded
constexpr size_t O_XHQ  = O_XHF + 2560;
constexpr size_t O_SYNC = O_XHQ + 2560;
constexpr size_t WS_NEED = O_SYNC + 256;

// ---------------- helpers ----------------
__device__ __forceinline__ unsigned short f32_bf16(float f) {
  unsigned u = __float_as_uint(f);
  u += 0x7FFFu + ((u >> 16) & 1u);      // round-to-nearest-even
  return (unsigned short)(u >> 16);
}

__device__ __forceinline__ float sigmoidf(float x) {
  return 1.0f / (1.0f + expf(-x));
}

// device-wide sense-reversing barrier (all NB blocks resident)
__device__ __forceinline__ void grid_sync(unsigned* sync) {
  __syncthreads();
  if (threadIdx.x == 0) {
    __threadfence();
    volatile unsigned* gen = sync + 1;
    unsigned g = *gen;
    unsigned arrived = atomicAdd(sync, 1u);
    if (arrived == (unsigned)(NB - 1)) {
      *(volatile unsigned*)sync = 0u;
      __threadfence();
      *gen = g + 1u;
    } else {
      while (*gen == g) { __builtin_amdgcn_s_sleep(1); }
    }
    __threadfence();
  }
  __syncthreads();
}

// D = W_tile(16xK) * v  via bf16 WMMA; vector replicated across B columns.
// pk: fragment-packed weights (tile-major, chunk-major, lane*16+j order)
// xh: bf16 vector; B fragment = contiguous 16 bf16 at chunk*32 (+16 for hi lanes)
__device__ __forceinline__ void wmma_matvec(const unsigned short* __restrict__ pk,
                                            const unsigned short* __restrict__ xh,
                                            float* __restrict__ outv,
                                            const float* __restrict__ b0,
                                            const float* __restrict__ b1,
                                            int tiles, int chunks,
                                            int wave, int lane) {
  for (int tile = wave; tile < tiles; tile += NWAVES) {
    const v16bf* A = reinterpret_cast<const v16bf*>(pk) + (size_t)tile * chunks * 32;
    if (tile + NWAVES < tiles)
      __builtin_prefetch((const void*)(pk + (size_t)(tile + NWAVES) * chunks * 512), 0, 1);
    v8f acc = {};
    for (int c = 0; c < chunks; ++c) {
      v16bf a = A[(size_t)c * 32 + lane];
      v16bf b = *reinterpret_cast<const v16bf*>(xh + c * 32 + (lane & 16));
      acc = __builtin_amdgcn_wmma_f32_16x16x32_bf16(
          false, a, false, b, (short)0, acc, false, false);
    }
    // lane0 holds rows tile*16+0..7 (all N columns equal), lane16 rows +8..15
    int rb = tile * 16 + ((lane & 16) ? 8 : 0);
    if ((lane & 15) == 0) {
#pragma unroll
      for (int m = 0; m < 8; ++m) {
        float v = acc[m];
        if (b0) v += b0[rb + m];
        if (b1) v += b1[rb + m];
        outv[rb + m] = v;
      }
    }
  }
}

// ---------------- weight -> bf16 WMMA fragment packer ----------------
// Builds A-fragments for W_cat = [Wa(rows x kA) | Wb(rows x kB)], K = kA+kB.
// A 16x32 bf16 layout (contiguous k): lanes<16 hold k {0..15}; lanes>=16 hold {16..31}.
__global__ void pack_frag(const float* __restrict__ Wa, const float* __restrict__ Wb,
                          unsigned short* __restrict__ out,
                          int kA, int kB, int tiles, int chunks) {
  size_t total = (size_t)tiles * chunks * 512;
  for (size_t idx = (size_t)blockIdx.x * blockDim.x + threadIdx.x; idx < total;
       idx += (size_t)gridDim.x * blockDim.x) {
    int j = (int)(idx & 15);
    int L = (int)((idx >> 4) & 31);
    size_t tc = idx >> 9;
    int c = (int)(tc % chunks);
    int tile = (int)(tc / chunks);
    int m = tile * 16 + (L & 15);
    int kk = c * 32 + (L & 16) + j;
    float v = (kk < kA) ? Wa[(size_t)m * kA + kk]
                        : Wb[(size_t)m * kB + (kk - kA)];
    out[idx] = f32_bf16(v);
  }
}

// ---------------- persistent kernel args ----------------
struct TKFArgs {
  const float *x, *y0, *P0, *H, *R, *Tl, *Tu;
  const float *bih_m, *bhh_m, *bfc_m;
  const float *bih_f, *bhh_f, *bfc_f;
  const float *bih_q, *bhh_q, *bfc_q;
  const unsigned short *pk_gm, *pk_gf, *pk_gq, *pk_fm, *pk_ff, *pk_fq;
  float *F;                 // f32 state region base
  unsigned short *xhm, *xhf, *xhq;
  unsigned *sync;
  float *out;
};

__global__ __launch_bounds__(NT)
void tkf_persistent(TKFArgs A) {
  const int tid = threadIdx.x;
  const int g   = blockIdx.x * blockDim.x + tid;
  const int lane = tid & 31;
  const int wave = g >> 5;

  float* YMID = A.F + F_YMID;
  float* HM = A.F + F_HM; float* CM = A.F + F_CM;
  float* HF = A.F + F_HF; float* CF = A.F + F_CF;
  float* HQ = A.F + F_HQ; float* CQ = A.F + F_CQ;
  float* GM = A.F + F_GM; float* GF = A.F + F_GF; float* GQ = A.F + F_GQ;
  float* FV = A.F + F_FV; float* QV = A.F + F_QV;
  float* P  = A.F + F_P;  float* R1 = A.F + F_R1; float* HP = A.F + F_HP;
  float* GJ = A.F + F_GJ; float* KB = A.F + F_KB;
  float* Zb = A.F + F_Z;  float* PUN = A.F + F_PUN;
  float* EY = A.F + F_EY; float* WV = A.F + F_WV;

  __shared__ float sM[64 * 192];
  __shared__ float sRow[192];
  __shared__ float sF[64];
  __shared__ float sPinv;

  // ---- init ----
  for (int i = g; i < 16384; i += NTHREADS) P[i] = A.P0[i];
  for (int i = g; i < 1024; i += NTHREADS) {
    HM[i] = CM[i] = HF[i] = CF[i] = HQ[i] = CQ[i] = 0.0f;
    A.xhm[STAN + i] = 0;
  }
  for (int i = g; i < 128; i += NTHREADS) A.xhm[i] = f32_bf16(A.y0[i]);
  grid_sync(A.sync);

  for (int t = 0; t < T_STEPS; ++t) {
    // ph1: g_m = [y|h_m] @ Wcat_m.T + biases   (WMMA, bf16)
    wmma_matvec(A.pk_gm, A.xhm, GM, A.bih_m, A.bhh_m, GATE_TILES, GATE_CHUNKS, wave, lane);
    grid_sync(A.sync);

    // ph2: LSTM-m cell elementwise; write h_m (f32 + bf16 into xhm tail)
    for (int j = g; j < HIDN; j += NTHREADS) {
      float I = sigmoidf(GM[j]);
      float Fg = sigmoidf(GM[HIDN + j]);
      float Gg = tanhf(GM[2 * HIDN + j]);
      float O = sigmoidf(GM[3 * HIDN + j]);
      float c = Fg * CM[j] + I * Gg;
      CM[j] = c;
      float h = O * tanhf(c);
      HM[j] = h;
      A.xhm[STAN + j] = f32_bf16(h);
    }
    grid_sync(A.sync);

    // ph3: y_mid = h_m @ Wfc_m.T + bfc_m (WMMA over 8 tiles)
    wmma_matvec(A.pk_fm, A.xhm + STAN, YMID, A.bfc_m, nullptr, FC_TILES, FC_CHUNKS, wave, lane);
    grid_sync(A.sync);

    // ph4: build xh_f / xh_q bf16; z = H @ y_mid
    for (int i = g; i < KCAT; i += NTHREADS) {
      unsigned short v = (i < STAN) ? f32_bf16(YMID[i])
                                    : 0;
      unsigned short vf = (i < STAN) ? v : f32_bf16(HF[i - STAN]);
      unsigned short vq = (i < STAN) ? v : f32_bf16(HQ[i - STAN]);
      A.xhf[i] = vf;
      A.xhq[i] = vq;
    }
    for (int o = g; o < OBSN; o += NTHREADS) {
      float s = 0.0f;
      for (int k = 0; k < STAN; ++k) s += A.H[o * STAN + k] * YMID[k];
      Zb[o] = s;
    }
    grid_sync(A.sync);

    // ph5: tobit scalars (threads 0..63) then g_f, g_q via WMMA
    if (g < OBSN) {
      int o = g;
      float rd = sqrtf(A.R[o * OBSN + o]) + 1e-4f;
      float zz = Zb[o];
      float zl = (A.Tl[o] - zz) / rd;
      float zu = (A.Tu[o] - zz) / rd;
      float cpl = 0.5f * (1.0f + erff(zl * 0.70710678118f));
      float cpu = 0.5f * (1.0f + erff(zu * 0.70710678118f));
      float ppl = 0.39894228040f * expf(-0.5f * zl * zl);
      float ppu = 0.39894228040f * expf(-0.5f * zu * zu);
      float p = cpu - cpl + 1e-4f;
      float l = (ppu - ppl) / p;
      float cc = zl * ppl - zu * ppu;
      PUN[o] = p;
      EY[o] = p * (zz - rd * l) + cpl * A.Tl[o] + (1.0f - cpu) * A.Tu[o];
      WV[o] = cc / p + 1.0f - l * l;
    }
    // 512 tiles: first 256 -> f, next 256 -> q
    for (int tile = wave; tile < 2 * GATE_TILES; tile += NWAVES) {
      const unsigned short* pk = (tile < GATE_TILES) ? A.pk_gf : A.pk_gq;
      const unsigned short* xh = (tile < GATE_TILES) ? A.xhf : A.xhq;
      float* outv = (tile < GATE_TILES) ? GF : GQ;
      const float* b0 = (tile < GATE_TILES) ? A.bih_f : A.bih_q;
      const float* b1 = (tile < GATE_TILES) ? A.bhh_f : A.bhh_q;
      int tt = (tile < GATE_TILES) ? tile : tile - GATE_TILES;
      const v16bf* Ap = reinterpret_cast<const v16bf*>(pk) + (size_t)tt * GATE_CHUNKS * 32;
      v8f acc = {};
      for (int c = 0; c < GATE_CHUNKS; ++c) {
        v16bf a = Ap[(size_t)c * 32 + lane];
        v16bf b = *reinterpret_cast<const v16bf*>(xh + c * 32 + (lane & 16));
        acc = __builtin_amdgcn_wmma_f32_16x16x32_bf16(
            false, a, false, b, (short)0, acc, false, false);
      }
      int rb = tt * 16 + ((lane & 16) ? 8 : 0);
      if ((lane & 15) == 0) {
#pragma unroll
        for (int m = 0; m < 8; ++m) outv[rb + m] = acc[m] + b0[rb + m] + b1[rb + m];
      }
    }
    grid_sync(A.sync);

    // ph6: LSTM-f and LSTM-q cells elementwise
    for (int idx = g; idx < 2 * HIDN; idx += NTHREADS) {
      bool isF = idx < HIDN;
      int j = isF ? idx : idx - HIDN;
      float* Gv = isF ? GF : GQ;
      float* Hv = isF ? HF : HQ;
      float* Cv = isF ? CF : CQ;
      unsigned short* xh = isF ? A.xhf : A.xhq;
      float I = sigmoidf(Gv[j]);
      float Fg = sigmoidf(Gv[HIDN + j]);
      float Gg = tanhf(Gv[2 * HIDN + j]);
      float O = sigmoidf(Gv[3 * HIDN + j]);
      float c = Fg * Cv[j] + I * Gg;
      Cv[j] = c;
      float h = O * tanhf(c);
      Hv[j] = h;
      xh[STAN + j] = f32_bf16(h);
    }
    grid_sync(A.sync);

    // ph7: Fv, Qv via WMMA (8+8 tiles)
    for (int tile = wave; tile < 2 * FC_TILES; tile += NWAVES) {
      bool isF = tile < FC_TILES;
      const unsigned short* pk = isF ? A.pk_ff : A.pk_fq;
      const unsigned short* xh = (isF ? A.xhf : A.xhq) + STAN;
      float* outv = isF ? FV : QV;
      const float* b0 = isF ? A.bfc_f : A.bfc_q;
      int tt = isF ? tile : tile - FC_TILES;
      const v16bf* Ap = reinterpret_cast<const v16bf*>(pk) + (size_t)tt * FC_CHUNKS * 32;
      v8f acc = {};
      for (int c = 0; c < FC_CHUNKS; ++c) {
        v16bf a = Ap[(size_t)c * 32 + lane];
        v16bf b = *reinterpret_cast<const v16bf*>(xh + c * 32 + (lane & 16));
        acc = __builtin_amdgcn_wmma_f32_16x16x32_bf16(
            false, a, false, b, (short)0, acc, false, false);
      }
      int rb = tt * 16 + ((lane & 16) ? 8 : 0);
      if ((lane & 15) == 0) {
#pragma unroll
        for (int m = 0; m < 8; ++m) outv[rb + m] = acc[m] + b0[rb + m];
      }
    }
    grid_sync(A.sync);

    // ph8: covariance predict  P = Fv P Fv^T + diag(Qv)
    for (int idx = g; idx < STAN * STAN; idx += NTHREADS) {
      int i = idx >> 7, j = idx & 127;
      float v = FV[i] * P[idx] * FV[j];
      if (i == j) v += QV[i];
      P[idx] = v;
    }
    grid_sync(A.sync);

    // ph9: R1 = (P H^T) * p ;  HP = H P
    for (int idx = g; idx < 2 * STAN * OBSN; idx += NTHREADS) {
      if (idx < STAN * OBSN) {
        int i = idx >> 6, b = idx & 63;
        float s = 0.0f;
        for (int k = 0; k < STAN; ++k) s += P[i * STAN + k] * A.H[b * STAN + k];
        R1[i * OBSN + b] = s * PUN[b];
      } else {
        int r = idx - STAN * OBSN;
        int o = r >> 7, j = r & 127;
        float s = 0.0f;
        for (int k = 0; k < STAN; ++k) s += A.H[o * STAN + k] * P[k * STAN + j];
        HP[o * STAN + j] = s;
      }
    }
    grid_sync(A.sync);

    // ph10: GJ = [R2^T | R1^T],  R2 = p (H R1) + R*wv(col)
    for (int idx = g; idx < OBSN * OBSN + STAN * OBSN; idx += NTHREADS) {
      if (idx < OBSN * OBSN) {
        int a = idx >> 6, b = idx & 63;
        float s = 0.0f;
        for (int k = 0; k < STAN; ++k) s += A.H[a * STAN + k] * R1[k * OBSN + b];
        float v = PUN[a] * s + A.R[a * OBSN + b] * WV[b];
        GJ[b * 192 + a] = v;                    // R2^T
      } else {
        int r = idx - OBSN * OBSN;
        int j = r >> 6, rr = r & 63;            // R1 is [STAN][OBSN]
        GJ[rr * 192 + 64 + j] = R1[j * OBSN + rr]; // R1^T
      }
    }
    grid_sync(A.sync);

    // ph11: block 0 Gauss-Jordan solve R2^T X = R1^T in LDS ; K = X^T
    if (blockIdx.x == 0) {
      for (int i = tid; i < 64 * 192; i += NT) sM[i] = GJ[i];
      __syncthreads();
      for (int r = 0; r < 64; ++r) {
        if (tid == 0) sPinv = 1.0f / sM[r * 192 + r];
        __syncthreads();
        if (tid < 192) sRow[tid] = sM[r * 192 + tid] * sPinv;
        else if (tid < 256) sF[tid - 192] = sM[(tid - 192) * 192 + r];
        __syncthreads();
        for (int i2 = tid; i2 < 64 * 192; i2 += NT) {
          int ri = i2 / 192, ci = i2 - ri * 192;
          sM[i2] = (ri == r) ? sRow[ci] : (sM[i2] - sF[ri] * sRow[ci]);
        }
        __syncthreads();
      }
      for (int i2 = tid; i2 < STAN * OBSN; i2 += NT) {
        int ii = i2 >> 6, oo = i2 & 63;
        KB[i2] = sM[oo * 192 + 64 + ii];        // K[i][o]
      }
    }
    grid_sync(A.sync);

    // ph12: state + covariance update, emit y_t
    const float* xt = A.x + (size_t)t * OBSN;
    for (int idx = g; idx < STAN + STAN * STAN; idx += NTHREADS) {
      if (idx < STAN) {
        int i = idx;
        float s = 0.0f;
        for (int o = 0; o < OBSN; ++o) s += KB[i * OBSN + o] * (xt[o] - EY[o]);
        float yn = YMID[i] + s;
        A.out[(size_t)t * STAN + i] = yn;
        A.xhm[i] = f32_bf16(yn);
      } else {
        int r = idx - STAN;
        int i = r >> 7, j = r & 127;
        float s = 0.0f;
        for (int o = 0; o < OBSN; ++o) s += KB[i * OBSN + o] * PUN[o] * HP[o * STAN + j];
        P[i * STAN + j] -= s;
      }
    }
    grid_sync(A.sync);
  }
}

// ---------------- host launcher ----------------
extern "C" void kernel_launch(void* const* d_in, const int* in_sizes, int n_in,
                              void* d_out, int out_size, void* d_ws, size_t ws_size,
                              hipStream_t stream) {
  (void)in_sizes; (void)n_in; (void)out_size;
  if (ws_size < WS_NEED) return;

  const float* x   = (const float*)d_in[0];
  const float* y0  = (const float*)d_in[1];
  const float* P0  = (const float*)d_in[2];
  const float* H   = (const float*)d_in[3];
  const float* R   = (const float*)d_in[4];
  const float* Tl  = (const float*)d_in[5];
  const float* Tu  = (const float*)d_in[6];
  const float* Wih_m = (const float*)d_in[7];
  const float* Whh_m = (const float*)d_in[8];
  const float* bih_m = (const float*)d_in[9];
  const float* bhh_m = (const float*)d_in[10];
  const float* Wfc_m = (const float*)d_in[11];
  const float* bfc_m = (const float*)d_in[12];
  const float* Wih_f = (const float*)d_in[13];
  const float* Whh_f = (const float*)d_in[14];
  const float* bih_f = (const float*)d_in[15];
  const float* bhh_f = (const float*)d_in[16];
  const float* Wfc_f = (const float*)d_in[17];
  const float* bfc_f = (const float*)d_in[18];
  const float* Wih_q = (const float*)d_in[19];
  const float* Whh_q = (const float*)d_in[20];
  const float* bih_q = (const float*)d_in[21];
  const float* bhh_q = (const float*)d_in[22];
  const float* Wfc_q = (const float*)d_in[23];
  const float* bfc_q = (const float*)d_in[24];

  char* ws = (char*)d_ws;
  unsigned short* pk_gm = (unsigned short*)(ws + O_PK_GM);
  unsigned short* pk_gf = (unsigned short*)(ws + O_PK_GF);
  unsigned short* pk_gq = (unsigned short*)(ws + O_PK_GQ);
  unsigned short* pk_fm = (unsigned short*)(ws + O_PK_FM);
  unsigned short* pk_ff = (unsigned short*)(ws + O_PK_FF);
  unsigned short* pk_fq = (unsigned short*)(ws + O_PK_FQ);

  // pack weights -> bf16 WMMA fragments (every launch; deterministic)
  pack_frag<<<2048, 256, 0, stream>>>(Wih_m, Whh_m, pk_gm, STAN, HIDN, GATE_TILES, GATE_CHUNKS);
  pack_frag<<<2048, 256, 0, stream>>>(Wih_f, Whh_f, pk_gf, STAN, HIDN, GATE_TILES, GATE_CHUNKS);
  pack_frag<<<2048, 256, 0, stream>>>(Wih_q, Whh_q, pk_gq, STAN, HIDN, GATE_TILES, GATE_CHUNKS);
  pack_frag<<<256, 256, 0, stream>>>(Wfc_m, Wfc_m, pk_fm, 0, HIDN, FC_TILES, FC_CHUNKS);
  pack_frag<<<256, 256, 0, stream>>>(Wfc_f, Wfc_f, pk_ff, 0, HIDN, FC_TILES, FC_CHUNKS);
  pack_frag<<<256, 256, 0, stream>>>(Wfc_q, Wfc_q, pk_fq, 0, HIDN, FC_TILES, FC_CHUNKS);

  hipMemsetAsync(ws + O_SYNC, 0, 256, stream);

  TKFArgs A;
  A.x = x; A.y0 = y0; A.P0 = P0; A.H = H; A.R = R; A.Tl = Tl; A.Tu = Tu;
  A.bih_m = bih_m; A.bhh_m = bhh_m; A.bfc_m = bfc_m;
  A.bih_f = bih_f; A.bhh_f = bhh_f; A.bfc_f = bfc_f;
  A.bih_q = bih_q; A.bhh_q = bhh_q; A.bfc_q = bfc_q;
  A.pk_gm = pk_gm; A.pk_gf = pk_gf; A.pk_gq = pk_gq;
  A.pk_fm = pk_fm; A.pk_ff = pk_ff; A.pk_fq = pk_fq;
  A.F = (float*)(ws + O_F32);
  A.xhm = (unsigned short*)(ws + O_XHM);
  A.xhf = (unsigned short*)(ws + O_XHF);
  A.xhq = (unsigned short*)(ws + O_XHQ);
  A.sync = (unsigned*)(ws + O_SYNC);
  A.out = (float*)d_out;

  tkf_persistent<<<NB, NT, 0, stream>>>(A);
}
